// MultiHeadAttention_19284403159232
// MI455X (gfx1250) — matrix operand order repaired and checked by reading, not compiled, not run on hardware
//
#include <hip/hip_runtime.h>
#include <stdint.h>

// Problem constants (match reference).
#define NB   4
#define SEQ  2048
#define HID  1024
#define NH   16
#define DK   64
#define MTOT (NB * SEQ)   // 8192 tokens

typedef unsigned short u16;
typedef __attribute__((ext_vector_type(16))) __bf16      v16bf;
typedef __attribute__((ext_vector_type(8)))  float        v8f;
typedef __attribute__((ext_vector_type(8)))  unsigned int v8u;

// ---------------------------------------------------------------------------
// Helpers
// ---------------------------------------------------------------------------

__device__ __forceinline__ u16 f2bf(float f) {
  unsigned u = __builtin_bit_cast(unsigned, f);
  u += 0x7FFFu + ((u >> 16) & 1u);            // round-to-nearest-even
  return (u16)(u >> 16);
}

// Two 16-byte loads composed into one 32-byte bf16 fragment (raw bits).
__device__ __forceinline__ v8u ld16x2(const u16* p0, const u16* p1) {
  uint4 a = *reinterpret_cast<const uint4*>(p0);
  uint4 b = *reinterpret_cast<const uint4*>(p1);
  v8u r;
  r[0] = a.x; r[1] = a.y; r[2] = a.z; r[3] = a.w;
  r[4] = b.x; r[5] = b.y; r[6] = b.z; r[7] = b.w;
  return r;
}

// A-fragment (16x32 bf16, M x K): lane m holds row (rowbase+m).
// ISA layout: lanes 0-15 carry K = k0+[0..7] and k0+[16..23];
//             lanes 16-31 carry K = k0+[8..15] and k0+[24..31].
__device__ __forceinline__ v8u frag_a(const u16* base, int ld, int rowbase,
                                      int k0, int lane) {
  const int m = lane & 15, hi = lane >> 4;
  const u16* p = base + (size_t)(rowbase + m) * ld + k0 + hi * 8;
  return ld16x2(p, p + 16);
}

// B-fragment (32x16 bf16, K x N): lane n holds column (colbase+n); each
// column's K values are contiguous in memory (stride ld per column).
// ISA layout: lanes 0-15 carry K = k0+[0..7] and k0+[16..23];
//             lanes 16-31 carry K = k0+[8..15] and k0+[24..31].
__device__ __forceinline__ v8u frag_b(const u16* base, int ld, int colbase,
                                      int k0, int lane) {
  const int m = lane & 15, hi = lane >> 4;
  const u16* p = base + (size_t)(colbase + m) * ld + k0 + hi * 8;
  return ld16x2(p, p + 16);
}

__device__ __forceinline__ v8f wmma_bf16(v8u a, v8u b, v8f c) {
  return __builtin_amdgcn_wmma_f32_16x16x32_bf16(
      false, __builtin_bit_cast(v16bf, a),
      false, __builtin_bit_cast(v16bf, b),
      (short)0, c, false, false);
}

__device__ __forceinline__ v8f zero8() {
  v8f z;
  #pragma unroll
  for (int i = 0; i < 8; ++i) z[i] = 0.0f;
  return z;
}

// gfx1250 async copy global -> LDS (16B per lane), tracked by ASYNCcnt.
__device__ __forceinline__ void async_g2l_b128(uint32_t lds_off, const u16* g) {
  asm volatile("global_load_async_to_lds_b128 %0, %1, off"
               :: "v"(lds_off), "v"((uint64_t)(uintptr_t)g)
               : "memory");
}
// Drain all outstanding async copies.
__device__ __forceinline__ void wait_async0() {
  asm volatile("s_wait_asynccnt 0x0" ::: "memory");
}
// Double-buffer drain: async loads retire in issue order per wave, so with 4
// newer copies in flight, ASYNCcnt <= 4 means the previous slab's 4 are done.
__device__ __forceinline__ void wait_async4() {
  asm volatile("s_wait_asynccnt 0x4" ::: "memory");
}
__device__ __forceinline__ uint32_t lds_addr_of(const void* p) {
  return (uint32_t)(uintptr_t)p;   // low 32 bits of generic ptr = LDS byte addr
}

// ---------------------------------------------------------------------------
// Kernel 1: fp32 -> bf16 cast (x and the four weight matrices)
// ---------------------------------------------------------------------------
__global__ void cast_bf16_kernel(const float* __restrict__ in,
                                 u16* __restrict__ out, int n) {
  int i = (blockIdx.x * blockDim.x + threadIdx.x) * 4;
  if (i + 3 < n) {
    const float4 f = *reinterpret_cast<const float4*>(in + i);
    uint2 o;
    o.x = (unsigned)f2bf(f.x) | ((unsigned)f2bf(f.y) << 16);
    o.y = (unsigned)f2bf(f.z) | ((unsigned)f2bf(f.w) << 16);
    *reinterpret_cast<uint2*>(out + i) = o;
  } else {
    for (; i < n; ++i) out[i] = f2bf(in[i]);
  }
}

// ---------------------------------------------------------------------------
// Kernel 2: WMMA GEMM  C[M x Ncol] = A_bf16[M x K] * W_bf16[Ncol x K]^T + bias
// WG = 256 thr = 8 waves (4 row-groups x 2 col-groups); WG tile 128x128;
// per-wave 32x64 (2x4 WMMA tiles); K-step 32.
// A and B K-slabs are DOUBLE-BUFFERED async copies into LDS (ASYNCcnt),
// overlapping DMA with WMMA; fragments read back as ds_load_b128.
// LDS rows padded to 80B to spread banks.
// mode 0: bf16 out, (N,H,S,dk) head layout   (Q, K)
// mode 1: bf16 out, (N,H,dk,S) transposed    (V — makes P@V B-frags contiguous)
// mode 2: fp32 out, row-major M x Ncol       (final projection -> d_out)
// ---------------------------------------------------------------------------
#define LPAD  40                 // padded LDS row stride in u16 (80B, 16B-aligned)
#define GBUF  (256 * LPAD)       // one buffer: 128 B-rows + 128 A-rows (20KB)

// Stage one 32-wide K-slab (B tile then A tile) into ldsbuf: 4 async/thread.
__device__ __forceinline__ void gemm_stage(uint32_t ldsbuf,
                                           const u16* __restrict__ W,
                                           const u16* __restrict__ A,
                                           int cbase, int rbase, int K, int kk,
                                           int tid) {
  #pragma unroll
  for (int c = 0; c < 2; ++c) {
    const int ch = tid * 2 + c;
    const int r = ch >> 2, part = ch & 3;
    async_g2l_b128(ldsbuf + (uint32_t)(r * LPAD * 2 + part * 16),
                   W + (size_t)(cbase + r) * K + kk + part * 8);
    async_g2l_b128(ldsbuf + (uint32_t)(128 * LPAD * 2 + r * LPAD * 2 + part * 16),
                   A + (size_t)(rbase + r) * K + kk + part * 8);
  }
}

__global__ __launch_bounds__(256) void gemm_wmma_kernel(
    const u16* __restrict__ A, const u16* __restrict__ W,
    const float* __restrict__ bias, u16* __restrict__ outb,
    float* __restrict__ outf, int K, int Ncol, int mode) {
  __shared__ u16 smem[2 * GBUF];   // 40KB, double-buffered

  const int tid  = threadIdx.x;
  const int lane = tid & 31;
  const int wid  = tid >> 5;
  const int m = lane & 15, hi = lane >> 4;
  const int rg = wid & 3;            // row group (32 rows each)
  const int cg = wid >> 2;           // col group (64 cols each)
  const int rbase = blockIdx.x * 128;
  const int cbase = blockIdx.y * 128;

  const uint32_t lds0 = lds_addr_of(smem);

  v8f acc[2][4];
  #pragma unroll
  for (int i = 0; i < 2; ++i)
    #pragma unroll
    for (int j = 0; j < 4; ++j) acc[i][j] = zero8();

  // Prologue: stage slab 0 into buffer 0.
  gemm_stage(lds0, W, A, cbase, rbase, K, 0, tid);

  for (int kk = 0; kk < K; kk += 32) {
    const int buf = (kk >> 5) & 1;
    if (kk + 32 < K) {
      // Issue next slab into the other buffer, then drain current slab only.
      gemm_stage(lds0 + (uint32_t)((buf ^ 1) * GBUF * 2), W, A, cbase, rbase,
                 K, kk + 32, tid);
      if (kk + 64 < K)  // warm GL2 two slabs ahead (speculative)
        __builtin_prefetch(A + (size_t)(rbase + (tid >> 1)) * K + kk + 64, 0, 1);
      wait_async4();
    } else {
      wait_async0();
    }
    __syncthreads();    // current slab visible to all waves

    const u16* sB = smem + buf * GBUF;
    const u16* sA = sB + 128 * LPAD;
    v8u a0 = frag_a(sA, LPAD, rg * 32,      0, lane);
    v8u a1 = frag_a(sA, LPAD, rg * 32 + 16, 0, lane);
    v8u b0 = frag_b(sB, LPAD, cg * 64,      0, lane);
    v8u b1 = frag_b(sB, LPAD, cg * 64 + 16, 0, lane);
    v8u b2 = frag_b(sB, LPAD, cg * 64 + 32, 0, lane);
    v8u b3 = frag_b(sB, LPAD, cg * 64 + 48, 0, lane);

    acc[0][0] = wmma_bf16(a0, b0, acc[0][0]);
    acc[0][1] = wmma_bf16(a0, b1, acc[0][1]);
    acc[0][2] = wmma_bf16(a0, b2, acc[0][2]);
    acc[0][3] = wmma_bf16(a0, b3, acc[0][3]);
    acc[1][0] = wmma_bf16(a1, b0, acc[1][0]);
    acc[1][1] = wmma_bf16(a1, b1, acc[1][1]);
    acc[1][2] = wmma_bf16(a1, b2, acc[1][2]);
    acc[1][3] = wmma_bf16(a1, b3, acc[1][3]);

    __syncthreads();    // reads done before this buffer is refilled next time
  }

  #pragma unroll
  for (int i = 0; i < 2; ++i) {
    #pragma unroll
    for (int jc = 0; jc < 4; ++jc) {
      const int col = cbase + cg * 64 + jc * 16 + m;
      const float bb = bias[col];
      #pragma unroll
      for (int jj = 0; jj < 8; ++jj) {
        // C layout: VGPR jj -> row jj (lanes 0-15) / jj+8 (lanes 16-31)
        const int row = rbase + rg * 32 + i * 16 + jj + hi * 8;
        const float val = acc[i][jc][jj] + bb;
        if (mode == 2) {
          outf[(size_t)row * Ncol + col] = val;
        } else {
          const int bn = row >> 11, s = row & (SEQ - 1);   // SEQ = 2048
          const int h  = col >> 6,  d = col & (DK - 1);    // DK  = 64
          const size_t addr =
              (mode == 0)
                  ? ((((size_t)bn * NH + h) * SEQ + s) * DK + d)
                  : ((((size_t)bn * NH + h) * DK + d) * SEQ + s);
          outb[addr] = f2bf(val);
        }
      }
    }
  }
}

// ---------------------------------------------------------------------------
// Kernel 3: flash attention per head.
// Grid (SEQ/64, NH, NB), block = 128 (4 waves, 16 q-rows each).
// The block's shared K tile (32x64) and V^T tile (64x32) per K-step are
// double-buffered async copies into LDS, consumed by all 4 waves (4x reuse).
// Online softmax fp32; P transposed C-layout -> A-layout through per-wave LDS
// with s_wait_dscnt; both matmuls via v_wmma_f32_16x16x32_bf16.
// ---------------------------------------------------------------------------
#define KPAD  72                       // K-tile row stride in elems (144B)
#define VPAD  40                       // V-tile row stride in elems (80B)
#define ABUF  (32 * KPAD + 64 * VPAD)  // one buffer: K tile + V tile (9.5KB)

// Stage K rows [kb0,kb0+32) and V^T slab for one K-step: 4 async/thread (128thr).
__device__ __forceinline__ void attn_stage(uint32_t ldsbuf,
                                           const u16* __restrict__ Kp,
                                           const u16* __restrict__ Vt,
                                           int kb0, int tid) {
  const uint32_t ldsV = ldsbuf + 32 * KPAD * 2;
  #pragma unroll
  for (int c = 0; c < 2; ++c) {        // K tile: 32 rows x 8 parts
    const int ch = tid * 2 + c;
    const int r = ch >> 3, part = ch & 7;
    async_g2l_b128(ldsbuf + (uint32_t)(r * KPAD * 2 + part * 16),
                   Kp + (size_t)(kb0 + r) * DK + part * 8);
  }
  #pragma unroll
  for (int c = 0; c < 2; ++c) {        // V tile: 64 rows x 4 parts
    const int ch = tid * 2 + c;
    const int r = ch >> 2, part = ch & 3;
    async_g2l_b128(ldsV + (uint32_t)(r * VPAD * 2 + part * 16),
                   Vt + (size_t)r * SEQ + kb0 + part * 8);
  }
}

__global__ __launch_bounds__(128) void attn_kernel(
    const u16* __restrict__ qb, const u16* __restrict__ kbuf,
    const u16* __restrict__ vtb, const int* __restrict__ mask,
    u16* __restrict__ ob) {
  // 2 x (K tile + V tile) double buffer + 4 per-wave 16x32 P tiles (~23KB).
  __shared__ u16 asmem[2 * ABUF + 4 * 512];

  const int tid  = threadIdx.x;
  const int lane = tid & 31;
  const int wid  = tid >> 5;
  const int m = lane & 15, hi = lane >> 4;
  const int bn = blockIdx.z, h = blockIdx.y;
  const int q0 = blockIdx.x * 64 + wid * 16;

  const size_t headoff = ((size_t)bn * NH + h) * (size_t)SEQ * DK;
  const u16* Q  = qb   + headoff;        // (S, dk) row-major
  const u16* Kp = kbuf + headoff;        // (S, dk) row-major
  const u16* Vt = vtb  + headoff;        // (dk, S) row-major (transposed V)
  const int* mk = mask + bn * SEQ;

  const uint32_t lds0 = lds_addr_of(asmem);
  u16* Pl = asmem + 2 * ABUF + wid * 512;

  // Q A-fragments for the full dk=64 (two K-slabs of 32), reused all loop.
  const v8u qa0 = frag_a(Q, DK, q0, 0,  lane);
  const v8u qa1 = frag_a(Q, DK, q0, 32, lane);

  float mrow[8], lrow[8];
  v8f o[4];
  #pragma unroll
  for (int j = 0; j < 8; ++j) { mrow[j] = -3.0e38f; lrow[j] = 0.0f; }
  #pragma unroll
  for (int t = 0; t < 4; ++t) o[t] = zero8();

  const float scale = 0.125f;  // 1/sqrt(dk)

  // Prologue: stage first K/V slab into buffer 0.
  attn_stage(lds0, Kp, Vt, 0, tid);

  for (int kb0 = 0; kb0 < SEQ; kb0 += 32) {
    const int buf = (kb0 >> 5) & 1;
    if (kb0 + 32 < SEQ) {
      attn_stage(lds0 + (uint32_t)((buf ^ 1) * ABUF * 2), Kp, Vt, kb0 + 32, tid);
      wait_async4();
    } else {
      wait_async0();
    }
    __syncthreads();   // current K/V tiles visible to all waves

    const u16* sK = asmem + buf * ABUF;
    const u16* sV = sK + 32 * KPAD;

    // Two 16x16 score tiles covering key columns [kb0, kb0+32)
    v8u kf00 = frag_b(sK, KPAD, 0,  0,  lane);
    v8u kf01 = frag_b(sK, KPAD, 0,  32, lane);
    v8u kf10 = frag_b(sK, KPAD, 16, 0,  lane);
    v8u kf11 = frag_b(sK, KPAD, 16, 32, lane);
    v8f s0 = zero8(), s1 = zero8();
    s0 = wmma_bf16(qa0, kf00, s0); s0 = wmma_bf16(qa1, kf01, s0);
    s1 = wmma_bf16(qa0, kf10, s1); s1 = wmma_bf16(qa1, kf11, s1);

    const float madd0 = mk[kb0 + m]      ? 0.0f : -1.0e30f;
    const float madd1 = mk[kb0 + 16 + m] ? 0.0f : -1.0e30f;

    #pragma unroll
    for (int j = 0; j < 8; ++j) {
      const float a0 = s0[j] * scale + madd0;
      const float a1 = s1[j] * scale + madd1;
      // row max over the 16 lanes of this half (xor masks stay in-half)
      float mx = fmaxf(a0, a1);
      #pragma unroll
      for (int off = 8; off > 0; off >>= 1)
        mx = fmaxf(mx, __shfl_xor(mx, off, 32));
      const float nm = fmaxf(mrow[j], mx);
      const float al = __expf(mrow[j] - nm);
      mrow[j] = nm;
      const float p0 = __expf(a0 - nm);
      const float p1 = __expf(a1 - nm);
      float rs = p0 + p1;
      #pragma unroll
      for (int off = 8; off > 0; off >>= 1)
        rs += __shfl_xor(rs, off, 32);
      lrow[j] = lrow[j] * al + rs;
      o[0][j] *= al; o[1][j] *= al; o[2][j] *= al; o[3][j] *= al;
      // scatter P (C layout) into row-major 16x32 per-wave LDS tile
      const int prow = j + hi * 8;
      Pl[prow * 32 + m]      = f2bf(p0);
      Pl[prow * 32 + 16 + m] = f2bf(p1);
    }

    // Make this wave's DS stores visible to its own DS loads.
    asm volatile("s_wait_dscnt 0x0" ::: "memory");

    // Re-read P as a 16x32 A-fragment.
    const v8u pa = ld16x2(Pl + m * 32 + hi * 8, Pl + m * 32 + 16 + hi * 8);

    // O(16x64) += P(16x32) @ V(32x64); V columns read from staged LDS tile.
    #pragma unroll
    for (int t = 0; t < 4; ++t) {
      v8u vf = frag_b(sV, VPAD, t * 16, 0, lane);
      o[t] = wmma_bf16(pa, vf, o[t]);
    }

    __syncthreads();   // reads done before this buffer is refilled next time
  }

  // Epilogue: normalize rows and emit context in (N,S,HID) bf16 layout.
  #pragma unroll
  for (int j = 0; j < 8; ++j) {
    const float inv = 1.0f / lrow[j];
    const int row = q0 + j + hi * 8;
    u16* orow = ob + ((size_t)bn * SEQ + row) * HID + h * DK;
    #pragma unroll
    for (int t = 0; t < 4; ++t)
      orow[t * 16 + m] = f2bf(o[t][j] * inv);
  }
}

// ---------------------------------------------------------------------------
// Host-side launcher
// ---------------------------------------------------------------------------
extern "C" void kernel_launch(void* const* d_in, const int* in_sizes, int n_in,
                              void* d_out, int out_size, void* d_ws,
                              size_t ws_size, hipStream_t stream) {
  const float* x    = (const float*)d_in[0];
  const int*   mask = (const int*)d_in[1];
  const float* Wq   = (const float*)d_in[2];
  const float* bq   = (const float*)d_in[3];
  const float* Wk   = (const float*)d_in[4];
  const float* bk   = (const float*)d_in[5];
  const float* Wv   = (const float*)d_in[6];
  const float* bv   = (const float*)d_in[7];
  const float* Wp   = (const float*)d_in[8];
  const float* bp   = (const float*)d_in[9];
  float* out = (float*)d_out;

  char* ws = (char*)d_ws;
  u16* xb  = (u16*)ws; ws += (size_t)MTOT * HID * 2;   // x bf16
  u16* wqb = (u16*)ws; ws += (size_t)HID * HID * 2;
  u16* wkb = (u16*)ws; ws += (size_t)HID * HID * 2;
  u16* wvb = (u16*)ws; ws += (size_t)HID * HID * 2;
  u16* wpb = (u16*)ws; ws += (size_t)HID * HID * 2;
  u16* qb  = (u16*)ws; ws += (size_t)MTOT * HID * 2;   // (N,H,S,dk)
  u16* kb  = (u16*)ws; ws += (size_t)MTOT * HID * 2;   // (N,H,S,dk)
  u16* vtb = (u16*)ws; ws += (size_t)MTOT * HID * 2;   // (N,H,dk,S)
  u16* ob  = (u16*)ws; ws += (size_t)MTOT * HID * 2;   // context (N,S,HID)

  const size_t nx = (size_t)MTOT * HID;
  const size_t nw = (size_t)HID * HID;
  cast_bf16_kernel<<<(int)((nx + 1023) / 1024), 256, 0, stream>>>(x, xb, (int)nx);
  cast_bf16_kernel<<<(int)((nw + 1023) / 1024), 256, 0, stream>>>(Wq, wqb, (int)nw);
  cast_bf16_kernel<<<(int)((nw + 1023) / 1024), 256, 0, stream>>>(Wk, wkb, (int)nw);
  cast_bf16_kernel<<<(int)((nw + 1023) / 1024), 256, 0, stream>>>(Wv, wvb, (int)nw);
  cast_bf16_kernel<<<(int)((nw + 1023) / 1024), 256, 0, stream>>>(Wp, wpb, (int)nw);

  dim3 ggrid(MTOT / 128, HID / 128);
  gemm_wmma_kernel<<<ggrid, 256, 0, stream>>>(xb, wqb, bq, qb,  nullptr, HID, HID, 0);
  gemm_wmma_kernel<<<ggrid, 256, 0, stream>>>(xb, wkb, bk, kb,  nullptr, HID, HID, 0);
  gemm_wmma_kernel<<<ggrid, 256, 0, stream>>>(xb, wvb, bv, vtb, nullptr, HID, HID, 1);

  dim3 agrid(SEQ / 64, NH, NB);
  attn_kernel<<<agrid, 128, 0, stream>>>(qb, kb, vtb, mask, ob);

  gemm_wmma_kernel<<<ggrid, 256, 0, stream>>>(ob, wpb, bp, nullptr, out, HID, HID, 2);
}
